// CannyEdgeDetector_73967926771786
// MI455X (gfx1250) — matrix operand order repaired and checked by reading, not compiled, not run on hardware
//
#include <hip/hip_runtime.h>

typedef __attribute__((ext_vector_type(16))) _Float16 v16h;
typedef __attribute__((ext_vector_type(8)))  float    v8f;

typedef int v4i_ __attribute__((vector_size(16)));
typedef __attribute__((address_space(1))) v4i_ gv4i;   // global (AS1) int4
typedef __attribute__((address_space(3))) v4i_ lv4i;   // LDS (AS3) int4

#if defined(__gfx1250__) && __has_builtin(__builtin_amdgcn_global_load_async_to_lds_b128)
#define HAS_ASYNC_LDS 1
#else
#define HAS_ASYNC_LDS 0
#endif

#define IMG_H 1024
#define IMG_W 1024
#define TILE  64

// Static LDS layout (bytes), regions aliased across pipeline stages:
//  [0      .. 29568)  IMGF  float[84][88]      -> later MAG  _Float16[66][67] (8844 B)
//  [29568  .. 45248)  TMPT  _Float16[80][98]   (transposed: [col][row]) -> later NMS _Float16[64][65]
//  [45248  .. 58208)  BLUR  _Float16[80][81]
#define SMEM_BYTES 58208

#define MGF(rr,cc) ((float)MAG[(rr)*67 + (cc)])

__global__ __launch_bounds__(256)
void canny_fused_kernel(const float* __restrict__ image,
                        unsigned int* __restrict__ weakbits,
                        unsigned int* __restrict__ strongbits)
{
    __shared__ char smem[SMEM_BYTES] __attribute__((aligned(16)));
    float*    IMGF = (float*)(smem);              // 84 x 88 (stage 1-2)
    _Float16* MAG  = (_Float16*)(smem);           // 66 x 67 (stage 4a+, aliases IMGF)
    _Float16* TMPT = (_Float16*)(smem + 29568);   // 80 cols x 98 rows, transposed (stage 2-3)
    _Float16* NMS  = (_Float16*)(smem + 29568);   // 64 x 65 (stage 4b+, aliases TMPT)
    _Float16* BLUR = (_Float16*)(smem + 45248);   // 80 x 81

    const int tid   = threadIdx.x;
    const int lane  = tid & 31;
    const int wave  = tid >> 5;
    const int hi    = (lane >= 16) ? 1 : 0;
    const int gx0   = blockIdx.x * TILE;
    const int gy0   = blockIdx.y * TILE;
    const int batch = blockIdx.z;
    const float* img = image + (size_t)batch * IMG_H * IMG_W;

    const float gk[5] = {0.05448868f, 0.24420135f, 0.40261996f,
                         0.24420135f, 0.05448868f};

    // Prefetch tile's image region into GL2 (global_prefetch_b8).
    if (tid < 64) {
        int py = gy0 - 10; if (py < 0) py = 0;
        int px = gx0 - 12; if (px < 0) px = 0;
        __builtin_prefetch(img + (size_t)py * IMG_W + px + tid * 64, 0, 0);
    }

    // ---- stage 1: stage 84x88 f32 halo tile into LDS ----
    // Interior tiles: async global->LDS DMA (ASYNCcnt), 16B-aligned b128 chunks.
    // Border tiles: guarded scalar loads with zero padding.
    const bool xin = (gx0 >= 12) && (gx0 + 76 <= IMG_W);
    const bool yin = (gy0 >= 10) && (gy0 + 74 <= IMG_H);
#if HAS_ASYNC_LDS
    if (xin && yin) {
        for (int i = tid; i < 84 * 22; i += 256) {
            int r = i / 22, ch = i - r * 22;
            const float* gsrc = img + (size_t)(gy0 - 10 + r) * IMG_W + (gx0 - 12) + 4 * ch;
            float* ldst = IMGF + r * 88 + 4 * ch;
            __builtin_amdgcn_global_load_async_to_lds_b128((gv4i*)gsrc, (lv4i*)ldst, 0, 0);
        }
        asm volatile("s_wait_asynccnt 0x0" ::: "memory");
    } else
#endif
    {
        for (int i = tid; i < 84 * 88; i += 256) {
            int r = i / 88, c = i - r * 88;
            int gy = gy0 - 10 + r, gx = gx0 - 12 + c;
            float v = 0.f;
            if (gy >= 0 && gy < IMG_H && gx >= 0 && gx < IMG_W)
                v = img[gy * IMG_W + gx];
            IMGF[r * 88 + c] = v;
        }
    }
    __syncthreads();

    // ---- stage 2: horizontal 5-tap gaussian -> TMPT transposed (84 rows x 80 cols) ----
    // tmp col c corresponds to image col gx0-8+c -> IMGF cols (c+2)..(c+6)
    for (int i = tid; i < 84 * 80; i += 256) {
        int t = i / 80, c = i - t * 80;
        const float* row = IMGF + t * 88 + c + 2;
        float acc = gk[0]*row[0] + gk[1]*row[1] + gk[2]*row[2]
                  + gk[3]*row[3] + gk[4]*row[4];
        TMPT[c * 98 + t] = (_Float16)acc;
    }
    // zero-pad rows 84..95 (read by WMMA B tiles with zero A weights)
    for (int i = tid; i < 80 * 12; i += 256) {
        int c = i / 12, t = 84 + (i - c * 12);
        TMPT[c * 98 + t] = (_Float16)0.f;
    }
    __syncthreads();

    // ---- A fragment: banded gaussian matrix G[16][32], G[m][k] = gk[k-m] for 0<=k-m<=4 ----
    v16h afrag;
    {
        int m = lane & 15;
#pragma unroll
        for (int v = 0; v < 8; ++v) {
            int base = (v < 4 ? 2 * v : 2 * v + 8) + (hi ? 8 : 0);
#pragma unroll
            for (int p = 0; p < 2; ++p) {
                int K = base + p;
                int d = K - m;
                float w = (d >= 0 && d <= 4) ? gk[d] : 0.f;
                afrag[2 * v + p] = (_Float16)w;
            }
        }
    }

    // ---- stage 3: vertical gaussian as WMMA banded matmul, 5x5 tiles of 16x16 ----
    for (int bi = wave; bi < 25; bi += 8) {
        int br = bi / 5;
        int blk_r = br * 16, blk_c = (bi - br * 5) * 16;
        int col = blk_c + (lane & 15);
        union { v16h h; unsigned int u[8]; } bu;
#pragma unroll
        for (int v = 0; v < 8; ++v) {
            int r0 = blk_r + (v < 4 ? 2 * v : 2 * v + 8) + (hi ? 8 : 0);
            bu.u[v] = *(const unsigned int*)(TMPT + col * 98 + r0); // rows r0,r0+1 packed
        }
        v8f acc = {};
        acc = __builtin_amdgcn_wmma_f32_16x16x32_f16(false, afrag, false, bu.h,
                                                     (short)0, acc, false, false);
#pragma unroll
        for (int v = 0; v < 8; ++v) {
            int row = blk_r + v + (hi ? 8 : 0);
            BLUR[row * 81 + col] = (_Float16)acc[v];
        }
    }
    __syncthreads();

    // ---- stage 4a: sobel magnitude on 66x66 (output + 1 halo), zero outside image ----
    for (int i = tid; i < 66 * 66; i += 256) {
        int r = i / 66, c = i - r * 66;
        int y = r + 7, x = c + 7;
        const _Float16* bp = BLUR + (y - 1) * 81 + (x - 1);
        float a00=(float)bp[0],   a01=(float)bp[1],   a02=(float)bp[2];
        float a10=(float)bp[81],                      a12=(float)bp[83];
        float a20=(float)bp[162], a21=(float)bp[163], a22=(float)bp[164];
        float gxv = (a02 + 2.f*a12 + a22) - (a00 + 2.f*a10 + a20);
        float gyv = (a20 + 2.f*a21 + a22) - (a00 + 2.f*a01 + a02);
        float m = sqrtf(gxv * gxv + gyv * gyv);
        int iy = gy0 + r - 1, ix = gx0 + c - 1;
        if (iy < 0 || iy >= IMG_H || ix < 0 || ix >= IMG_W) m = 0.f;
        MAG[r * 67 + c] = (_Float16)m;
    }
    __syncthreads();

    // ---- stage 4b: non-maximum suppression on 64x64 outputs ----
    for (int i = tid; i < 64 * 64; i += 256) {
        int oy = i >> 6, ox = i & 63;
        int y = oy + 8, x = ox + 8;
        const _Float16* bp = BLUR + (y - 1) * 81 + (x - 1);
        float a00=(float)bp[0],   a01=(float)bp[1],   a02=(float)bp[2];
        float a10=(float)bp[81],                      a12=(float)bp[83];
        float a20=(float)bp[162], a21=(float)bp[163], a22=(float)bp[164];
        float gxv = (a02 + 2.f*a12 + a22) - (a00 + 2.f*a10 + a20);
        float gyv = (a20 + 2.f*a21 + a22) - (a00 + 2.f*a01 + a02);
        float m = MGF(oy + 1, ox + 1);
        float ax = fabsf(gxv), ay = fabsf(gyv);
        float p, q;
        if (ay < 0.41421356f * ax) {          // E / W
            p = MGF(oy + 1, ox + 2); q = MGF(oy + 1, ox);
        } else if (ay >= 2.41421356f * ax) {  // N / S
            p = MGF(oy,     ox + 1); q = MGF(oy + 2, ox + 1);
        } else if (gxv * gyv > 0.f) {         // NE / SW
            p = MGF(oy,     ox + 2); q = MGF(oy + 2, ox);
        } else {                               // NW / SE
            p = MGF(oy,     ox);     q = MGF(oy + 2, ox + 2);
        }
        float nv = (m >= p && m >= q) ? m : 0.f;
        NMS[oy * 65 + ox] = (_Float16)nv;
    }
    __syncthreads();

    // ---- stage 5: threshold + bit-pack weak/strong masks (32 px per u32 word) ----
    if (tid < 128) {
        int row = tid >> 1, half = tid & 1;
        unsigned int wk = 0u, st = 0u;
        const _Float16* np = NMS + row * 65 + half * 32;
#pragma unroll
        for (int b = 0; b < 32; ++b) {
            float v = (float)np[b];
            wk |= (v >= 0.1f ? 1u : 0u) << b;
            st |= (v >= 0.2f ? 1u : 0u) << b;
        }
        int gw = ((batch << 10) + gy0 + row) * 32 + (gx0 >> 5) + half;
        weakbits[gw]   = wk;
        strongbits[gw] = st;
    }
}

// One hysteresis iteration: 3x3 binary dilation of strong mask, AND weak mask.
__global__ __launch_bounds__(256)
void hyst_dilate_kernel(const unsigned int* __restrict__ sin_,
                        const unsigned int* __restrict__ weak,
                        unsigned int* __restrict__ sout)
{
    int idx = blockIdx.x * 256 + threadIdx.x;     // 0 .. 8*1024*32 - 1
    int im  = idx >> 15;
    int r   = (idx >> 5) & 1023;
    int c   = idx & 31;
    const unsigned int* base = sin_ + (im << 15);
    unsigned int acc = 0u;
#pragma unroll
    for (int dy = -1; dy <= 1; ++dy) {
        int rr = r + dy;
        if (rr < 0 || rr >= 1024) continue;
        const unsigned int* rowp = base + (rr << 5);
        unsigned int cn = rowp[c];
        unsigned int lf = (c > 0)  ? rowp[c - 1] : 0u;
        unsigned int rt = (c < 31) ? rowp[c + 1] : 0u;
        acc |= cn | (cn << 1) | (cn >> 1) | (lf >> 31) | (rt << 31);
    }
    sout[idx] = acc & weak[idx];
}

// Expand bit mask to f32 {0,1} output.
__global__ __launch_bounds__(256)
void expand_kernel(const unsigned int* __restrict__ sbits,
                   float* __restrict__ out)
{
    int idx = blockIdx.x * 256 + threadIdx.x;
    unsigned int w = sbits[idx >> 5];
    out[idx] = ((w >> (idx & 31)) & 1u) ? 1.0f : 0.0f;
}

extern "C" void kernel_launch(void* const* d_in, const int* in_sizes, int n_in,
                              void* d_out, int out_size, void* d_ws, size_t ws_size,
                              hipStream_t stream)
{
    const float* image = (const float*)d_in[0];
    float* out = (float*)d_out;

    const int WORDS = 8 * 1024 * 32;              // bit-packed words per mask
    unsigned int* weak = (unsigned int*)d_ws;     // 1 MB
    unsigned int* sA   = weak + WORDS;            // 1 MB (strong ping)
    unsigned int* sB   = sA + WORDS;              // 1 MB (strong pong)

    dim3 g1(16, 16, 8);
    canny_fused_kernel<<<g1, 256, 0, stream>>>(image, weak, sA);

    unsigned int* pin = sA;
    unsigned int* pout = sB;
    for (int it = 0; it < 16; ++it) {
        hyst_dilate_kernel<<<WORDS / 256, 256, 0, stream>>>(pin, weak, pout);
        unsigned int* t = pin; pin = pout; pout = t;
    }
    expand_kernel<<<(8 * 1024 * 1024) / 256, 256, 0, stream>>>(pin, out);
}
